// SoftRGCNConv_30966714204816
// MI455X (gfx1250) — matrix operand order repaired and checked by reading, not compiled, not run on hardware
//
#include <hip/hip_runtime.h>

// Problem constants (match reference)
#define Bq 8
#define Nq 1024
#define Iq 256
#define Oq 256
#define Rq 8
#define KKq (Nq * Rq)   // 8192 : flattened contraction dim k = n*R + r

typedef __attribute__((ext_vector_type(16))) _Float16     v16h;
typedef __attribute__((ext_vector_type(8)))  float        v8f;
typedef __attribute__((ext_vector_type(4)))  float        f32x4;
typedef __attribute__((ext_vector_type(4)))  unsigned int u32x4;
typedef __attribute__((ext_vector_type(4)))  int          i32x4;

// Async-copy builtin wants non-const int4 pointers with explicit addr spaces:
//   param0: int4 addrspace(1)* (global source)   [prints as "int4 __device__*"]
//   param1: int4 addrspace(3)* (LDS destination)
#define GPTR(p) ((__attribute__((address_space(1))) i32x4*)(uintptr_t)(p))
#define LPTR(p) ((__attribute__((address_space(3))) i32x4*)(uintptr_t)(p))

static __device__ __forceinline__ unsigned pack2h(float lo, float hi) {
  union { _Float16 h[2]; unsigned u; } v;
  v.h[0] = (_Float16)lo;
  v.h[1] = (_Float16)hi;
  return v.u;
}

// ---------------------------------------------------------------------------
// Kernel 0: transpose weight [R][I][O] f32  ->  wT [R][O][I] f16
// ---------------------------------------------------------------------------
__global__ void k_wT(const float* __restrict__ w, _Float16* __restrict__ wT) {
  int tid = blockIdx.x * 256 + threadIdx.x;
  int r   = tid / (Oq * Iq);
  int rem = tid % (Oq * Iq);
  int o   = rem / Iq;
  int i   = rem % Iq;
  wT[tid] = (_Float16)w[((size_t)r * Iq + i) * Oq + o];
}

// ---------------------------------------------------------------------------
// Kernel 1: features  f[b,n,r,o] = sum_i x[b,n,i] * W[r,i,o]
// One wave = one (16n x 16o) tile for ALL 8 relations (8 f32 accumulators).
// Output layout fT[b][o][k], k = n*R + r (f16): each lane stores 128
// contiguous bytes; the aggregation kernel's B-operand becomes contiguous.
// ---------------------------------------------------------------------------
__global__ void __launch_bounds__(128) k_feat(const float* __restrict__ x,
                                              const _Float16* __restrict__ wT,
                                              _Float16* __restrict__ fT) {
  const int wave = threadIdx.x >> 5;
  const int lane = threadIdx.x & 31;
  const int lm   = lane & 15;
  const int hf   = lane >> 4;

  const int tile = blockIdx.x * 4 + wave;      // 8192 tiles total
  const int b    = tile >> 10;                 // 1024 tiles per batch
  const int nt   = (tile >> 4) & 63;
  const int ot   = tile & 15;
  const int n0   = nt * 16;
  const int o0   = ot * 16;

  const float*    xrow = x  + ((size_t)(b * Nq + n0 + lm)) * Iq + 16 * hf;
  const _Float16* wcol = wT + ((size_t)(o0 + lm)) * Iq + 16 * hf;

  v8f acc[Rq] = {};

  for (int i0 = 0; i0 < Iq; i0 += 32) {
    f32x4 xa0 = *(const f32x4*)(xrow + i0);
    f32x4 xa1 = *(const f32x4*)(xrow + i0 + 4);
    f32x4 xb0 = *(const f32x4*)(xrow + i0 + 8);
    f32x4 xb1 = *(const f32x4*)(xrow + i0 + 12);
    v16h a;
#pragma unroll
    for (int e = 0; e < 4; ++e) {
      a[e]      = (_Float16)xa0[e];
      a[e + 4]  = (_Float16)xa1[e];
      a[e + 8]  = (_Float16)xb0[e];
      a[e + 12] = (_Float16)xb1[e];
    }
#pragma unroll
    for (int r = 0; r < Rq; ++r) {
      v16h bm = *(const v16h*)(wcol + (size_t)r * (Oq * Iq) + i0);
      acc[r] = __builtin_amdgcn_wmma_f32_16x16x32_f16(
          false, a, false, bm, (short)0, acc[r], false, false);
    }
  }

  // Pack this lane's 8n x 8r block (64 f16 = 128 contiguous bytes) and store.
  unsigned words[32];
#pragma unroll
  for (int j = 0; j < 8; ++j)
#pragma unroll
    for (int rp = 0; rp < 4; ++rp)
      words[j * 4 + rp] = pack2h(acc[2 * rp][j], acc[2 * rp + 1][j]);

  _Float16* dst =
      fT + ((size_t)(b * Oq + o0 + lm)) * KKq + (size_t)(n0 + 8 * hf) * Rq;
  u32x4* d4 = (u32x4*)dst;
#pragma unroll
  for (int q = 0; q < 8; ++q) {
    u32x4 v = {words[4 * q], words[4 * q + 1], words[4 * q + 2],
               words[4 * q + 3]};
    d4[q] = v;
  }
}

// ---------------------------------------------------------------------------
// Kernel 2: aggregation  out[b,m,o] = (sum_k A[k,m] * F[k,o]) / denom[b,m]
// Workgroup = one m-strip (16 rows) x full O; wave w covers o in [w*64,(w+1)*64).
// adj A-slabs (128 k = 16 source rows, each 512 B contiguous) are staged into
// LDS with GLOBAL_LOAD_ASYNC_TO_LDS_B128, double-buffered on ASYNCcnt, and
// shared by all 4 waves -> adj streamed from HBM exactly once, no per-wave
// redundant vmem. B-operand (fT, f16, L2-resident) loads stay direct.
// ---------------------------------------------------------------------------
#define SLABK 128   // k per slab = 16 source-node rows
#define ROWF  128   // floats per row (16 m * 8 r)
#define ROWP  132   // padded LDS row stride (floats) to spread banks

__global__ void __launch_bounds__(128) k_agg(const float* __restrict__ adj,
                                             const _Float16* __restrict__ fT,
                                             float* __restrict__ out) {
  __shared__ float slab[2][16 * ROWP];

  const int wave = threadIdx.x >> 5;
  const int lane = threadIdx.x & 31;
  const int lm   = lane & 15;
  const int hf   = lane >> 4;

  const int b  = blockIdx.x >> 6;
  const int m0 = (blockIdx.x & 63) * 16;

  // adj element [b, n, m, r] at b*N*N*R + n*N*R + m*R + r ; for fixed n the
  // 16m x 8r block starting at m0 is 512 contiguous bytes.
  const float* adjbase = adj + (size_t)b * Nq * Nq * Rq + (size_t)m0 * Rq;
  const _Float16* fcol =
      fT + ((size_t)(b * Oq + wave * 64 + lm)) * KKq + 16 * hf;

  v8f   acc[4] = {};
  float dsum   = 0.f;

  const int S = KKq / SLABK;  // 64 slabs

  // Each wave async-copies 4 rows of the slab (one b128 per lane per row).
  auto issue = [&](int bufidx, int s) {
#pragma unroll
    for (int q = 0; q < 4; ++q) {
      const int    row = 4 * wave + q;
      const float* g =
          adjbase + (size_t)(16 * s + row) * (Nq * Rq) + lane * 4;
      float* l = &slab[bufidx][row * ROWP + lane * 4];
      __builtin_amdgcn_global_load_async_to_lds_b128(GPTR(g), LPTR(l), 0, 0);
    }
  };

  issue(0, 0);  // prologue fill

  for (int s = 0; s < S; ++s) {
    const int cur = s & 1;
    if (s + 1 < S) {
      issue(1 - cur, s + 1);
      __builtin_amdgcn_s_wait_asynccnt(4);  // slab s retired (in-order)
    } else {
      __builtin_amdgcn_s_wait_asynccnt(0);
    }
    __syncthreads();  // slab[cur] visible to all waves

#pragma unroll
    for (int t = 0; t < 4; ++t) {
      const int    n1 = 4 * t + 2 * hf;
      const int    n2 = 4 * t + 2 * hf + 1;
      const float* L1 = &slab[cur][n1 * ROWP + lm * 8];
      const float* L2 = &slab[cur][n2 * ROWP + lm * 8];
      f32x4 A0 = *(const f32x4*)L1;
      f32x4 A1 = *(const f32x4*)(L1 + 4);
      f32x4 A2 = *(const f32x4*)L2;
      f32x4 A3 = *(const f32x4*)(L2 + 4);

      v16h a;
#pragma unroll
      for (int e = 0; e < 4; ++e) {
        a[e]      = (_Float16)A0[e];
        a[e + 4]  = (_Float16)A1[e];
        a[e + 8]  = (_Float16)A2[e];
        a[e + 12] = (_Float16)A3[e];
        dsum += A0[e] + A1[e] + A2[e] + A3[e];
      }

      const int kk = s * SLABK + 32 * t;
#pragma unroll
      for (int u = 0; u < 4; ++u) {
        v16h bm = *(const v16h*)(fcol + (size_t)u * 16 * KKq + kk);
        acc[u] = __builtin_amdgcn_wmma_f32_16x16x32_f16(
            false, a, false, bm, (short)0, acc[u], false, false);
      }
    }
    __syncthreads();  // all waves done reading slab[cur] before reuse
  }

  // Combine lane halves: lane L and L+16 hold disjoint K ranges of row m0+lm.
  dsum += __shfl_xor(dsum, 16, 32);
  float inv[8];
#pragma unroll
  for (int j = 0; j < 8; ++j) {
    float dv = __shfl(dsum, 8 * hf + j, 32);  // denom for row m0 + 8*hf + j
    inv[j]   = 1.f / fmaxf(dv, 1e-16f);
  }

  float* obase =
      out + ((size_t)b * Nq + m0 + 8 * hf) * Oq + wave * 64 + lm;
#pragma unroll
  for (int t = 0; t < 4; ++t)
#pragma unroll
    for (int j = 0; j < 8; ++j)
      obase[(size_t)j * Oq + t * 16] = acc[t][j] * inv[j];
}

// ---------------------------------------------------------------------------
// Launch: ws = wT (1 MB f16) + fT (33.5 MB f16), both fully overwritten
// before use each call (no cross-call state).
// ---------------------------------------------------------------------------
extern "C" void kernel_launch(void* const* d_in, const int* in_sizes, int n_in,
                              void* d_out, int out_size, void* d_ws,
                              size_t ws_size, hipStream_t stream) {
  const float* x   = (const float*)d_in[0];
  const float* adj = (const float*)d_in[1];
  const float* w   = (const float*)d_in[2];
  float*       out = (float*)d_out;

  _Float16* wT = (_Float16*)d_ws;                    // R*O*I   f16
  _Float16* fT = wT + (size_t)Rq * Oq * Iq;          // B*O*KK  f16

  k_wT<<<(Rq * Oq * Iq) / 256, 256, 0, stream>>>(w, wT);
  k_feat<<<(Bq * (Nq / 16) * (Oq / 16)) / 4, 128, 0, stream>>>(x, wT, fT);
  k_agg<<<Bq * (Nq / 16), 128, 0, stream>>>(adj, fT, out);
}
